// local_neighbourhood_cluster_73890617360811
// MI455X (gfx1250) — matrix operand order repaired and checked by reading, not compiled, not run on hardware
//
#include <hip/hip_runtime.h>
#include <hip/hip_bf16.h>
#include <math.h>

#define NN 200000
#define EE 500000
#define BB 2000
#define DD 100

#define WPAD 112                 // padded N (7*16)
#define KPAD_D 128               // padded K for D-weights (4*32)
#define KPAD_E 224               // padded K for We (7*32)
#define WSZ_D (KPAD_D * WPAD)    // 14336 bf16 per D-weight
#define WSZ_E (KPAD_E * WPAD)    // 25088 bf16 for We

typedef __bf16 v16bf __attribute__((ext_vector_type(16)));
typedef float  v8f   __attribute__((ext_vector_type(8)));

__device__ __forceinline__ v8f wmma_bf16(v16bf a, v16bf b, v8f c) {
  return __builtin_amdgcn_wmma_f32_16x16x32_bf16(false, a, false, b, (short)0, c, false, false);
}

// B fragment from pre-packed, zero-padded bf16 weights: lane%16 = N col, elements = K rows in table order.
__device__ __forceinline__ v16bf wload_bp(const __bf16* __restrict__ Wb, int k0, int n0, int lane) {
  v16bf b;
  int n = n0 + (lane & 15);
  int kb = (lane < 16) ? 0 : 8;
#pragma unroll
  for (int e = 0; e < 16; ++e)
    b[e] = Wb[(long long)(k0 + kb + ((e < 8) ? 0 : 16) + (e & 7)) * WPAD + n];
  return b;
}

// monotone float<->uint encoding so atomicMax(uint) implements float max
__device__ __forceinline__ unsigned enc_f32(float f) {
  unsigned u = __float_as_uint(f);
  return (u & 0x80000000u) ? ~u : (u | 0x80000000u);
}
__device__ __forceinline__ float dec_f32(unsigned u) {
  u = (u & 0x80000000u) ? (u & 0x7FFFFFFFu) : ~u;
  return __uint_as_float(u);
}

// ---------------- pass 0: init segment max/sum ----------------
__global__ void init_kernel(unsigned* __restrict__ smax, float* __restrict__ ssum) {
  int idx = blockIdx.x * blockDim.x + threadIdx.x;
  if (idx < 2 * NN) {
    smax[idx] = 0x007FFFFFu; // enc(-inf)
    ssum[idx] = 0.0f;
  }
}

// ---------------- pass 0b: pack weights to zero-padded bf16, pack biases ----------------
__global__ void prep_kernel(const float* __restrict__ Wq, const float* __restrict__ Wk,
                            const float* __restrict__ Wv, const float* __restrict__ Wskip,
                            const float* __restrict__ We,
                            const float* __restrict__ bq, const float* __restrict__ bk,
                            const float* __restrict__ bv, const float* __restrict__ bskip,
                            __bf16* __restrict__ Wbf, float* __restrict__ biasp)
{
  int idx = blockIdx.x * blockDim.x + threadIdx.x;
  const int tot4 = 4 * WSZ_D;
  if (idx < tot4) {
    int w = idx / WSZ_D, rem = idx % WSZ_D;
    int r = rem / WPAD, c = rem % WPAD;
    const float* W = (w == 0) ? Wq : (w == 1) ? Wk : (w == 2) ? Wv : Wskip;
    Wbf[idx] = (__bf16)((r < DD && c < DD) ? W[r * DD + c] : 0.0f);
  } else if (idx < tot4 + WSZ_E) {
    int rem = idx - tot4;
    int r = rem / WPAD, c = rem % WPAD;
    Wbf[idx] = (__bf16)((r < 2 * DD && c < DD) ? We[r * DD + c] : 0.0f);
  } else if (idx < tot4 + WSZ_E + 4 * WPAD) {
    int rem = idx - tot4 - WSZ_E;
    int w = rem / WPAD, c = rem % WPAD;
    const float* bsrc = (w == 0) ? bq : (w == 1) ? bk : (w == 2) ? bv : bskip;
    biasp[rem] = (c < DD) ? bsrc[c] : 0.0f;
  }
}

// ---------------- pass 1: q,k,v,skip = z @ W + b (bf16 WMMA) ----------------
// One wave per (16-row tile, matrix). All pointers = single base + uniform offset -> global loads.
__global__ void __launch_bounds__(128) node_gemm_kernel(
    const float* __restrict__ z, const __bf16* __restrict__ Wbf,
    const float* __restrict__ biasp, float* __restrict__ qbase)
{
  int wave = threadIdx.x >> 5;
  int lane = threadIdx.x & 31;
  int row0 = blockIdx.x * 16;

  const __bf16* W    = Wbf + wave * WSZ_D;
  const float*  bias = biasp + wave * WPAD;
  float*        dst  = qbase + (long long)wave * ((long long)NN * DD);

  int m  = lane & 15;
  int kb = (lane < 16) ? 0 : 8;
  const float* zrow = z + (long long)(row0 + m) * DD;

  v8f acc[7] = {};
  for (int kt = 0; kt < 4; ++kt) {           // K = 100 -> 4 tiles of 32 (A zero-padded)
    int k0 = kt * 32;
    v16bf a;
#pragma unroll
    for (int e = 0; e < 16; ++e) {
      int kk = k0 + kb + ((e < 8) ? 0 : 16) + (e & 7);
      a[e] = (__bf16)((kk < DD) ? zrow[kk] : 0.0f);
    }
#pragma unroll
    for (int nt = 0; nt < 7; ++nt) {         // N = 100 -> 7 tiles of 16
      acc[nt] = wmma_bf16(a, wload_bp(W, k0, nt * 16, lane), acc[nt]);
    }
  }

  int rbase = (lane < 16) ? 0 : 8;
#pragma unroll
  for (int nt = 0; nt < 7; ++nt) {
    int col = nt * 16 + m;
    if (col < DD) {
      float bv_ = bias[col];
#pragma unroll
      for (int r = 0; r < 8; ++r)
        dst[(long long)(row0 + rbase + r) * DD + col] = acc[nt][r] + bv_;
    }
  }
}

// ---------------- pass 2: edge_attr GEMM + scores + ve (fused) ----------------
__global__ void __launch_bounds__(128) edge_attn_kernel(
    const float* __restrict__ last_update, const float* __restrict__ t,
    const float* __restrict__ msg,
    const int* __restrict__ edge_src, const int* __restrict__ edge_dst,
    const float* __restrict__ time_w, const float* __restrict__ time_b,
    const __bf16* __restrict__ Webf,
    const float* __restrict__ qbuf, const float* __restrict__ kbuf,
    const float* __restrict__ vbuf,
    float* __restrict__ vebuf, float* __restrict__ scorebuf,
    unsigned* __restrict__ segmax)
{
  __shared__ float smem[4][16][WPAD];
  const int ntiles = EE / 16;                 // 31250, exact
  int wave = threadIdx.x >> 5;
  int lane = threadIdx.x & 31;
  int tile = blockIdx.x * 4 + wave;
  bool valid = (tile < ntiles);
  if (!valid) tile = ntiles - 1;              // wave-uniform clamp: EXEC stays all-1s

  int e0 = tile * 16;
  int m  = lane & 15;
  int kb = (lane < 16) ? 0 : 8;
  int ge = e0 + m;
  int es = edge_src[ge];
  float rel = last_update[es] - t[ge];
  const float* msgrow = msg + (long long)ge * DD;

  v8f acc[7] = {};
  for (int kt = 0; kt < 7; ++kt) {            // K = 200 -> 7 tiles of 32 (A zero-padded)
    int k0 = kt * 32;
    v16bf a;
#pragma unroll
    for (int e = 0; e < 16; ++e) {
      int f = k0 + kb + ((e < 8) ? 0 : 16) + (e & 7);
      float val;
      if (f < DD)           val = cosf(rel * time_w[f] + time_b[f]);  // time encoding
      else if (f < 2 * DD)  val = msgrow[f - DD];                     // msg features
      else                  val = 0.0f;
      a[e] = (__bf16)val;
    }
#pragma unroll
    for (int nt = 0; nt < 7; ++nt) {
      acc[nt] = wmma_bf16(a, wload_bp(Webf, k0, nt * 16, lane), acc[nt]);
    }
  }

  // spill 16x112 edge_attr tile to LDS
  int rbase = (lane < 16) ? 0 : 8;
#pragma unroll
  for (int nt = 0; nt < 7; ++nt) {
    int col = nt * 16 + m;
#pragma unroll
    for (int r = 0; r < 8; ++r) smem[wave][rbase + r][col] = acc[nt][r];
  }
  __syncthreads();

  // attention score: lane -> (edge = lane&15, head = lane>>4); score = q[dst].(k[src]+ea)/sqrt(50)
  {
    int er = lane & 15;
    int h  = lane >> 4;
    int g  = e0 + er;
    int dn = edge_dst[g];
    int sn = edge_src[g];
    const float* qrow = qbuf + (long long)dn * DD + h * 50;
    const float* krow = kbuf + (long long)sn * DD + h * 50;
    float s = 0.0f;
    for (int j = 0; j < 50; ++j)
      s += qrow[j] * (krow[j] + smem[wave][er][h * 50 + j]);
    s *= 0.14142135623730951f;                // 1/sqrt(50)
    if (valid) {
      scorebuf[(long long)g * 2 + h] = s;
      atomicMax(&segmax[dn * 2 + h], enc_f32(s));
    }
  }

  // ve = v[src] + edge_attr, stored for the aggregation pass
  if (valid) {
    for (int idx = lane; idx < 16 * DD; idx += 32) {
      int er = idx / DD, d = idx % DD;
      int g  = e0 + er;
      int sn = edge_src[g];
      vebuf[(long long)g * DD + d] = smem[wave][er][d] + vbuf[(long long)sn * DD + d];
    }
  }
}

// ---------------- pass 3: ex = exp(score - max), seg_sum += ex ----------------
__global__ void softmax_exp_kernel(float* __restrict__ score, const int* __restrict__ edge_dst,
                                   const unsigned* __restrict__ smax, float* __restrict__ ssum)
{
  int idx = blockIdx.x * blockDim.x + threadIdx.x;
  if (idx >= 2 * EE) return;
  int e = idx >> 1, h = idx & 1;
  int dn = edge_dst[e];
  float ex = expf(score[idx] - dec_f32(smax[dn * 2 + h]));
  score[idx] = ex;
  atomicAdd(&ssum[dn * 2 + h], ex);
}

// ---------------- pass 4: out[dst] += alpha * ve ----------------
__global__ void aggregate_kernel(const float* __restrict__ vebuf, const float* __restrict__ ex,
                                 const float* __restrict__ ssum, const int* __restrict__ edge_dst,
                                 float* __restrict__ outz)
{
  long long gid = (long long)blockIdx.x * blockDim.x + threadIdx.x;
  if (gid >= (long long)EE * DD) return;
  int e = (int)(gid / DD);
  int d = (int)(gid % DD);
  int h = d / 50;
  int dn = edge_dst[e];
  float alpha = ex[(long long)e * 2 + h] / ssum[dn * 2 + h];
  atomicAdd(&outz[(long long)dn * DD + d], alpha * vebuf[gid]);
}

// ---------------- pass 5: link predictor ----------------
__global__ void __launch_bounds__(128) linkpred_kernel(
    const float* __restrict__ outz,
    const int* __restrict__ src, const int* __restrict__ pos_dst, const int* __restrict__ neg_dst,
    const int* __restrict__ adj_src, const int* __restrict__ adj_dst, const int* __restrict__ adj_neg,
    const float* __restrict__ Ws, const float* __restrict__ Wd,
    const float* __restrict__ Was, const float* __restrict__ Wad,
    const float* __restrict__ bh, const float* __restrict__ Wf, const float* __restrict__ bf,
    float* __restrict__ out)
{
  __shared__ float zs[DD], zd[DD], za[DD], zb2[DD];
  __shared__ float red[128];
  int row = blockIdx.x;
  bool isneg = (row >= BB);
  int b = isneg ? row - BB : row;
  int tid = threadIdx.x;
  int sn = src[b];
  int dn = isneg ? neg_dst[b] : pos_dst[b];
  const int* adjb = isneg ? adj_neg : adj_dst;

  for (int d = tid; d < DD; d += 128) {
    zs[d] = outz[(long long)sn * DD + d];
    zd[d] = outz[(long long)dn * DD + d];
    float a = 0.0f, c = 0.0f;
    for (int j = 0; j < 3; ++j) {
      a += outz[(long long)adj_src[b * 3 + j] * DD + d];
      c += outz[(long long)adjb[b * 3 + j] * DD + d];
    }
    za[d]  = a * (1.0f / 3.0f);
    zb2[d] = c * (1.0f / 3.0f);
  }
  __syncthreads();

  float partial = 0.0f;
  for (int d = tid; d < DD; d += 128) {
    float h = bh[d];
    for (int j = 0; j < DD; ++j) {
      h += zs[j] * Ws[j * DD + d] + zd[j] * Wd[j * DD + d]
         + za[j] * Was[j * DD + d] + zb2[j] * Wad[j * DD + d];
    }
    partial += fmaxf(h, 0.0f) * Wf[d];
  }
  red[tid] = partial;
  __syncthreads();
  for (int s2 = 64; s2 > 0; s2 >>= 1) {
    if (tid < s2) red[tid] += red[tid + s2];
    __syncthreads();
  }
  if (tid == 0) out[row] = red[0] + bf[0];
}

extern "C" void kernel_launch(void* const* d_in, const int* in_sizes, int n_in,
                              void* d_out, int out_size, void* d_ws, size_t ws_size,
                              hipStream_t stream)
{
  (void)in_sizes; (void)n_in; (void)out_size; (void)ws_size;
  const float* z      = (const float*)d_in[0];
  const float* lastup = (const float*)d_in[1];
  const float* t      = (const float*)d_in[2];
  const float* msg    = (const float*)d_in[3];
  const int* edge_src = (const int*)d_in[4];
  const int* edge_dst = (const int*)d_in[5];
  const int* src      = (const int*)d_in[6];
  const int* pos_dst  = (const int*)d_in[7];
  const int* neg_dst  = (const int*)d_in[8];
  const int* adj_src  = (const int*)d_in[9];
  const int* adj_dst  = (const int*)d_in[10];
  const int* adj_neg  = (const int*)d_in[11];
  const float* time_w = (const float*)d_in[12];
  const float* time_b = (const float*)d_in[13];
  const float* Wq     = (const float*)d_in[14];
  const float* bq     = (const float*)d_in[15];
  const float* Wk     = (const float*)d_in[16];
  const float* bk     = (const float*)d_in[17];
  const float* Wv     = (const float*)d_in[18];
  const float* bv     = (const float*)d_in[19];
  const float* We     = (const float*)d_in[20];
  const float* Wskip  = (const float*)d_in[21];
  const float* bskip  = (const float*)d_in[22];
  const float* Ws     = (const float*)d_in[23];
  const float* Wd     = (const float*)d_in[24];
  const float* Was    = (const float*)d_in[25];
  const float* Wad    = (const float*)d_in[26];
  const float* bh     = (const float*)d_in[27];
  const float* Wf     = (const float*)d_in[28];
  const float* bf     = (const float*)d_in[29];
  float* out = (float*)d_out;

  // workspace layout: [bf16 packed weights | packed biases | q,k,v,outz | ve | score | ssum | smax]
  char* base = (char*)d_ws;
  const int wbf_elems = 4 * WSZ_D + WSZ_E;          // 82432 bf16
  __bf16* Wbf  = (__bf16*)base;                      // 164864 B (32B-aligned rows)
  float* biasp = (float*)(base + 164864);            // 448 floats = 1792 B
  float* fbase = (float*)(base + 166656);

  const long long ND = (long long)NN * DD;
  const long long ED = (long long)EE * DD;
  float* qbuf  = fbase;
  float* kbuf  = qbuf + ND;
  float* vbuf  = kbuf + ND;
  float* outz  = vbuf + ND;
  float* vebuf = outz + ND;
  float* score = vebuf + ED;
  float* ssum  = score + 2LL * EE;
  unsigned* smax = (unsigned*)(ssum + 2LL * NN);

  const __bf16* Webf = Wbf + 4 * WSZ_D;

  init_kernel<<<(2 * NN + 255) / 256, 256, 0, stream>>>(smax, ssum);
  prep_kernel<<<(wbf_elems + 4 * WPAD + 255) / 256, 256, 0, stream>>>(
      Wq, Wk, Wv, Wskip, We, bq, bk, bv, bskip, Wbf, biasp);
  node_gemm_kernel<<<NN / 16, 128, 0, stream>>>(z, Wbf, biasp, qbuf);
  edge_attn_kernel<<<(EE / 16 + 3) / 4, 128, 0, stream>>>(lastup, t, msg, edge_src, edge_dst,
                                                          time_w, time_b, Webf,
                                                          qbuf, kbuf, vbuf,
                                                          vebuf, score, smax);
  softmax_exp_kernel<<<(2 * EE + 255) / 256, 256, 0, stream>>>(score, edge_dst, smax, ssum);
  aggregate_kernel<<<(int)((ED + 255) / 256), 256, 0, stream>>>(vebuf, score, ssum, edge_dst, outz);
  linkpred_kernel<<<2 * BB, 128, 0, stream>>>(outz, src, pos_dst, neg_dst,
                                              adj_src, adj_dst, adj_neg,
                                              Ws, Wd, Was, Wad, bh, Wf, bf, out);
}
